// Controller_sequence_48464410968741
// MI455X (gfx1250) — compile-verified, matrix operand order checked by reading, not run on hardware
//
#include <hip/hip_runtime.h>
#include <math.h>

#define NUM_OP     2048
#define NUM_GATES  (4*NUM_OP)      // 8192
#define NUM_ACTION 1024
#define NUM_BATCH  32
#define KCHUNKS    (NUM_OP/32)     // 64 k-chunks of 32 for 16x16x32 bf16 WMMA
#define MTILES     (NUM_GATES/16)  // 512 row tiles of 16
#define GEMV_WAVES 8               // waves per block, K split 8 ways

typedef __attribute__((ext_vector_type(16))) __bf16 v16bf;
typedef __attribute__((ext_vector_type(8)))  float  v8f;

// fp32 -> bf16, round-to-nearest-even
__device__ __forceinline__ unsigned short f2bf(float f) {
    unsigned int u = __float_as_uint(f);
    u += 0x7fffu + ((u >> 16) & 1u);
    return (unsigned short)(u >> 16);
}

// ---------------------------------------------------------------------------
// Zero the recurrent state (d_ws is poisoned once by the harness and never
// re-poisoned, so this must run every call).
// ---------------------------------------------------------------------------
__global__ void init_state(float* __restrict__ h, float* __restrict__ c,
                           unsigned short* __restrict__ hbf) {
    int j = blockIdx.x * blockDim.x + threadIdx.x;
    if (j < NUM_OP) { h[j] = 0.f; c[j] = 0.f; hbf[j] = 0; }
}

// ---------------------------------------------------------------------------
// Pack W_hh (fp32, 8192x2048 row-major) into per-wave bf16 A-fragments laid
// out exactly as V_WMMA_F32_16X16X32_BF16 expects:
//   lane L (M = L&15, half = L>>4) holds 16 bf16:
//     e in [0,8):  K = half*8 + e
//     e in [8,16): K = 16 + half*8 + (e-8)
// Fragment for (mt,kt) tile is 32 lanes x 32 bytes, contiguous.
// ---------------------------------------------------------------------------
__global__ void pack_whh(const float* __restrict__ W, unsigned short* __restrict__ Wp) {
    int gid  = blockIdx.x * blockDim.x + threadIdx.x;  // 512*64*32 threads
    int lane = gid & 31;
    int tile = gid >> 5;               // mt*64 + kt
    int kt   = tile & (KCHUNKS - 1);
    int mt   = tile >> 6;
    int half = lane >> 4;
    int m    = mt * 16 + (lane & 15);
    const float* row = W + (size_t)m * NUM_OP + kt * 32;

    union { unsigned short s[16]; uint4 q[2]; } frag;
#pragma unroll
    for (int e = 0; e < 8; ++e) frag.s[e]     = f2bf(row[half * 8 + e]);
#pragma unroll
    for (int e = 0; e < 8; ++e) frag.s[8 + e] = f2bf(row[16 + half * 8 + e]);

    uint4* dst = (uint4*)(Wp + ((size_t)tile * 32 + lane) * 16);
    dst[0] = frag.q[0];
    dst[1] = frag.q[1];
}

// ---------------------------------------------------------------------------
// x_gates[m] = (1/NUM_OP) * rowsum(W_ih[m]) + b_ih[m] + b_hh[m]
// One wave (32 lanes) per output row.
// ---------------------------------------------------------------------------
__global__ void xgates_kernel(const float* __restrict__ Wih, const float* __restrict__ bih,
                              const float* __restrict__ bhh, float* __restrict__ xg) {
    int wave = threadIdx.x >> 5;
    int lane = threadIdx.x & 31;
    int row  = blockIdx.x * (blockDim.x >> 5) + wave;
    const float* r = Wih + (size_t)row * NUM_OP;
    float s = 0.f;
    for (int k = lane; k < NUM_OP; k += 32) s += r[k];
#pragma unroll
    for (int off = 16; off; off >>= 1) s += __shfl_xor(s, off, 32);
    if (lane == 0) xg[row] = s * (1.0f / NUM_OP) + bih[row] + bhh[row];
}

// ---------------------------------------------------------------------------
// One LSTM step GEMV: g = W_hh @ h + x_gates via v_wmma_f32_16x16x32_bf16.
// One block (8 waves) per 16-row tile; K (64 chunks) split 8 per wave to
// shorten the dependent-WMMA chain; B = h broadcast into all 16 columns,
// staged in LDS; partial D tiles reduced through LDS.
// ---------------------------------------------------------------------------
__global__ void lstm_gemv_wmma(const unsigned short* __restrict__ Wp,
                               const unsigned short* __restrict__ hbf,
                               const float* __restrict__ xg,
                               float* __restrict__ g) {
    __shared__ __align__(16) unsigned short hs[NUM_OP];   // 4 KB bf16 h
    __shared__ float part[GEMV_WAVES][16];

    int tid  = threadIdx.x;
    int lane = tid & 31;
    int wave = tid >> 5;
    int mt   = blockIdx.x;
    int half = lane >> 4;

    // stage h (bf16) into LDS: 256 threads x 16B = 4 KB
    ((uint4*)hs)[tid] = ((const uint4*)hbf)[tid];
    __syncthreads();

    v8f acc = {};
    const uint4* abase = (const uint4*)(Wp + ((size_t)mt * KCHUNKS) * 32 * 16);
    int kt0 = wave * (KCHUNKS / GEMV_WAVES);

#pragma unroll
    for (int i = 0; i < KCHUNKS / GEMV_WAVES; ++i) {
        int kt = kt0 + i;
        union { v16bf v; uint4 q[2]; } A, B;
        const uint4* ap = abase + ((size_t)kt * 32 + lane) * 2;
        A.q[0] = ap[0];
        A.q[1] = ap[1];
        // B[k, n] = h[k] for all n: lane needs h[k0 + half*8 .. +7] and
        // h[k0 + 16 + half*8 .. +7] per the 16-bit B fragment mapping.
        B.q[0] = *(const uint4*)(hs + kt * 32 + half * 8);
        B.q[1] = *(const uint4*)(hs + kt * 32 + 16 + half * 8);
        acc = __builtin_amdgcn_wmma_f32_16x16x32_bf16(
            /*neg_a=*/false, A.v, /*neg_b=*/false, B.v,
            /*c_mod=*/(short)0, acc, /*reuse_a=*/false, /*reuse_b=*/false);
    }

    // D layout: VGPR r -> M=r (lanes 0-15), M=r+8 (lanes 16-31), N = lane&15.
    if ((lane & 15) == 0) {
#pragma unroll
        for (int r = 0; r < 8; ++r) part[wave][half * 8 + r] = acc[r];
    }
    __syncthreads();

    if (tid < 16) {
        float s = 0.f;
#pragma unroll
        for (int w = 0; w < GEMV_WAVES; ++w) s += part[w][tid];
        int m = mt * 16 + tid;
        g[m] = s + xg[m];
    }
}

// ---------------------------------------------------------------------------
// Gate nonlinearities + state update; also emits bf16 h for the next step's
// WMMA B-operand and records h_t for the softmax epilogue.
// ---------------------------------------------------------------------------
__global__ void lstm_activation(const float* __restrict__ g, float* __restrict__ h,
                                float* __restrict__ c, unsigned short* __restrict__ hbf,
                                float* __restrict__ outputs, int t) {
    int j = blockIdx.x * blockDim.x + threadIdx.x;
    float ig = 1.f / (1.f + __expf(-g[j]));
    float fg = 1.f / (1.f + __expf(-g[NUM_OP + j]));
    float gg = tanhf(g[2 * NUM_OP + j]);
    float og = 1.f / (1.f + __expf(-g[3 * NUM_OP + j]));
    float cn = fg * c[j] + ig * gg;
    float hn = og * tanhf(cn);
    c[j] = cn;
    h[j] = hn;
    hbf[j] = f2bf(hn);
    outputs[(size_t)t * NUM_OP + j] = hn;
}

// ---------------------------------------------------------------------------
// Per-timestep epilogue: logsumexp, entropy contribution, REINFORCE gather.
// Writes ent[t] and rf[t] (no float atomics -> deterministic accumulation).
// ---------------------------------------------------------------------------
__global__ void post_kernel(const float* __restrict__ outputs,
                            const long long* __restrict__ action,
                            const float* __restrict__ rewards,
                            float* __restrict__ ent, float* __restrict__ rf) {
    int t = blockIdx.x;
    const float* x = outputs + (size_t)t * NUM_OP;
    __shared__ float red[8];
    __shared__ float bcast;
    int tid = threadIdx.x, lane = tid & 31, wave = tid >> 5;

    // row max
    float m = -INFINITY;
    for (int j = tid; j < NUM_OP; j += blockDim.x) m = fmaxf(m, x[j]);
#pragma unroll
    for (int off = 16; off; off >>= 1) m = fmaxf(m, __shfl_xor(m, off, 32));
    if (lane == 0) red[wave] = m;
    __syncthreads();
    if (tid == 0) {
        float mm = red[0];
        for (int w = 1; w < 8; ++w) mm = fmaxf(mm, red[w]);
        bcast = mm;
    }
    __syncthreads();
    m = bcast;
    __syncthreads();

    // sum exp
    float s = 0.f;
    for (int j = tid; j < NUM_OP; j += blockDim.x) s += __expf(x[j] - m);
#pragma unroll
    for (int off = 16; off; off >>= 1) s += __shfl_xor(s, off, 32);
    if (lane == 0) red[wave] = s;
    __syncthreads();
    if (tid == 0) {
        float ss = 0.f;
        for (int w = 0; w < 8; ++w) ss += red[w];
        bcast = m + __logf(ss);
    }
    __syncthreads();
    float lse = bcast;
    __syncthreads();

    // entropy_t = -sum_j p*logp
    float e = 0.f;
    for (int j = tid; j < NUM_OP; j += blockDim.x) {
        float lp = x[j] - lse;
        e += __expf(lp) * lp;
    }
#pragma unroll
    for (int off = 16; off; off >>= 1) e += __shfl_xor(e, off, 32);
    if (lane == 0) red[wave] = e;
    __syncthreads();
    if (tid == 0) {
        float ee = 0.f;
        for (int w = 0; w < 8; ++w) ee += red[w];
        ent[t] = -ee;
    }

    // gather: action is int64 (32 x 1024 row-major)
    if (tid < NUM_BATCH) {
        long long a = action[(size_t)tid * NUM_ACTION + t];
        float contrib = (x[(int)a] - lse) * rewards[tid];
#pragma unroll
        for (int off = 16; off; off >>= 1) contrib += __shfl_xor(contrib, off, 32);
        if (tid == 0) rf[t] = contrib;
    }
}

// ---------------------------------------------------------------------------
// Deterministic fixed-order reduction of the 1024 per-timestep partials.
// ---------------------------------------------------------------------------
__global__ void finalize_kernel(const float* __restrict__ ent, const float* __restrict__ rf,
                                float* __restrict__ out) {
    __shared__ float se[256], sr[256];
    int tid = threadIdx.x;
    float a = 0.f, b = 0.f;
    for (int i = tid; i < NUM_ACTION; i += 256) { a += ent[i]; b += rf[i]; }
    se[tid] = a; sr[tid] = b;
    __syncthreads();
    for (int s = 128; s; s >>= 1) {
        if (tid < s) { se[tid] += se[tid + s]; sr[tid] += sr[tid + s]; }
        __syncthreads();
    }
    if (tid == 0) {
        // reinforce = -sr[0]; entropy_sum = se[0]
        out[0] = (-sr[0] + (float)NUM_BATCH * se[0]) / (float)NUM_BATCH;
    }
}

// ---------------------------------------------------------------------------
extern "C" void kernel_launch(void* const* d_in, const int* in_sizes, int n_in,
                              void* d_out, int out_size, void* d_ws, size_t ws_size,
                              hipStream_t stream) {
    (void)in_sizes; (void)n_in; (void)out_size; (void)ws_size;

    const float*     W_ih    = (const float*)d_in[0];
    const float*     W_hh    = (const float*)d_in[1];
    const float*     b_ih    = (const float*)d_in[2];
    const float*     b_hh    = (const float*)d_in[3];
    const float*     rewards = (const float*)d_in[4];
    const long long* action  = (const long long*)d_in[5];  // jnp.int64

    char* ws = (char*)d_ws;
    unsigned short* Wp      = (unsigned short*)(ws + 0);          // 32 MiB packed bf16
    float*          outputs = (float*)(ws + 33554432);            // 8 MiB h_t history
    float*          xg      = (float*)(ws + 41943040);            // 8192 f32
    float*          g       = (float*)(ws + 41975808);            // 8192 f32
    float*          h       = (float*)(ws + 42008576);            // 2048 f32
    float*          c       = (float*)(ws + 42016768);            // 2048 f32
    unsigned short* hbf     = (unsigned short*)(ws + 42024960);   // 2048 bf16
    float*          ent     = (float*)(ws + 42029056);            // 1024 f32
    float*          rf      = (float*)(ws + 42033152);            // 1024 f32

    init_state<<<(NUM_OP + 255) / 256, 256, 0, stream>>>(h, c, hbf);
    pack_whh<<<(MTILES * KCHUNKS * 32) / 256, 256, 0, stream>>>(W_hh, Wp);
    xgates_kernel<<<NUM_GATES / 8, 256, 0, stream>>>(W_ih, b_ih, b_hh, xg);

    for (int t = 0; t < NUM_ACTION; ++t) {
        lstm_gemv_wmma<<<MTILES, 32 * GEMV_WAVES, 0, stream>>>(Wp, hbf, xg, g);
        lstm_activation<<<NUM_OP / 256, 256, 0, stream>>>(g, h, c, hbf, outputs, t);
    }

    post_kernel<<<NUM_ACTION, 256, 0, stream>>>(outputs, action, rewards, ent, rf);
    finalize_kernel<<<1, 256, 0, stream>>>(ent, rf, (float*)d_out);
}
